// Attractor_2662879723819
// MI455X (gfx1250) — matrix operand order repaired and checked by reading, not compiled, not run on hardware
//
#include <hip/hip_runtime.h>
#include <hip/hip_bf16.h>

typedef __attribute__((ext_vector_type(16))) __bf16 v16bf;
typedef __attribute__((ext_vector_type(8)))  float  v8f;
typedef __attribute__((ext_vector_type(8)))  int    v8i;
typedef __attribute__((ext_vector_type(4)))  int    v4i;

#define CC 256
#define NN 512
#define KITER 15
#define LDS_STRIDE 520   // 512 + 8 bf16 pad -> 4-bank rotation per row

__device__ __forceinline__ unsigned int f2bf_bits(float f) {
  unsigned int u = __float_as_uint(f);
  return (u + 0x7FFFu + ((u >> 16) & 1u)) >> 16;   // round-to-nearest-even
}

__device__ __forceinline__ float fast_tanh(float x) {
  x = fminf(fmaxf(x, -15.0f), 15.0f);
  float e = __expf(2.0f * x);
  return (e - 1.0f) / (e + 1.0f);
}

// ---- weight fragment prep: pack into WMMA B-fragment layout (bf16) ----
// B frag (32x16, bf16): lane = {khalf(1), n(4)}; lanes 0-15: K=0..15, lanes 16-31: K=16..31
// vgpr v holds K pair (khalf*16 + 2v, +1). Frag = 32 lanes * 8 dwords = 1KB.

__global__ void prep_ws(const float* __restrict__ W, unsigned int* __restrict__ out) {
  int tid  = blockIdx.x * 256 + threadIdx.x;       // 131072 threads
  int v    = tid & 7;
  int lane = (tid >> 3) & 31;
  int frag = tid >> 8;                             // kt*32 + nt  (KT=16, NT=32)
  int nt = frag & 31, kt = frag >> 5;
  int n  = nt * 16 + (lane & 15);
  int k0 = kt * 32 + (lane >> 4) * 16 + v * 2;
  // B[k][n] = Ws[n][k], Ws = 0.5*(W + W^T)
  float w0 = 0.5f * (W[(size_t)n * NN + k0]     + W[(size_t)k0 * NN + n]);
  float w1 = 0.5f * (W[(size_t)n * NN + k0 + 1] + W[(size_t)(k0 + 1) * NN + n]);
  out[frag * 256 + lane * 8 + v] = f2bf_bits(w0) | (f2bf_bits(w1) << 16);
}

__global__ void prep_win(const float* __restrict__ Win, unsigned int* __restrict__ out) {
  int tid  = blockIdx.x * 256 + threadIdx.x;       // 65536 threads
  int v    = tid & 7;
  int lane = (tid >> 3) & 31;
  int frag = tid >> 8;                             // kt*32 + nt  (KT=8, NT=32)
  int nt = frag & 31, kt = frag >> 5;
  int n  = nt * 16 + (lane & 15);
  int k0 = kt * 32 + (lane >> 4) * 16 + v * 2;
  // B[k][n] = W_in[n][k], W_in is (512,256)
  float w0 = Win[(size_t)n * CC + k0];
  float w1 = Win[(size_t)n * CC + k0 + 1];
  out[frag * 256 + lane * 8 + v] = f2bf_bits(w0) | (f2bf_bits(w1) << 16);
}

__global__ void prep_wout(const float* __restrict__ Wout, unsigned int* __restrict__ out) {
  int tid  = blockIdx.x * 256 + threadIdx.x;       // 65536 threads
  int v    = tid & 7;
  int lane = (tid >> 3) & 31;
  int frag = tid >> 8;                             // kt*16 + nt  (KT=16, NT=16)
  int nt = frag & 15, kt = frag >> 4;
  int n  = nt * 16 + (lane & 15);
  int k0 = kt * 32 + (lane >> 4) * 16 + v * 2;
  // B[k][n] = W_out[n][k], W_out is (256,512)
  float w0 = Wout[(size_t)n * NN + k0];
  float w1 = Wout[(size_t)n * NN + k0 + 1];
  out[frag * 256 + lane * 8 + v] = f2bf_bits(w0) | (f2bf_bits(w1) << 16);
}

// ---- fragment loaders ----
// A frag (16x32 bf16, lane = {khalf, m}): vgprs 0-7 hold K=khalf*16+0..15
// -> two contiguous 16B chunks in row-major bf16 LDS: offsets (k0) and (k0+8)*2 bytes.
__device__ __forceinline__ v16bf load_afrag_lds(const unsigned short* lds, int byte_off) {
  v4i a = *(const v4i*)((const char*)lds + byte_off);
  v4i b = *(const v4i*)((const char*)lds + byte_off + 16);
  v8i r;
  r[0] = a[0]; r[1] = a[1]; r[2] = a[2]; r[3] = a[3];
  r[4] = b[0]; r[5] = b[1]; r[6] = b[2]; r[7] = b[3];
  return __builtin_bit_cast(v16bf, r);
}

__device__ __forceinline__ v16bf load_bfrag(const unsigned int* fragbase, int lane) {
  const v4i* p = (const v4i*)(fragbase + lane * 8);
  v4i a = p[0];
  v4i b = p[1];
  v8i r;
  r[0] = a[0]; r[1] = a[1]; r[2] = a[2]; r[3] = a[3];
  r[4] = b[0]; r[5] = b[1]; r[6] = b[2]; r[7] = b[3];
  return __builtin_bit_cast(v16bf, r);
}

#define WMMA_BF16(A, B, C) \
  __builtin_amdgcn_wmma_f32_16x16x32_bf16(false, (A), false, (B), (short)0, (C), false, false)

// ---- fused attractor kernel: 32 rows per workgroup, 8 waves ----
__global__ __launch_bounds__(256) void attractor_main(
    const float* __restrict__ x, const float* __restrict__ b_in,
    const float* __restrict__ brec, const float* __restrict__ b_out,
    const unsigned int* __restrict__ Wsf, const unsigned int* __restrict__ Winf,
    const unsigned int* __restrict__ Woutf, float* __restrict__ y) {
  __shared__ __attribute__((aligned(16))) unsigned short aLDS[32 * LDS_STRIDE];

  const int t     = threadIdx.x;
  const int wave  = t >> 5;
  const int lane  = t & 31;
  const int m     = lane & 15;
  const int khalf = lane >> 4;
  const long row_base = (long)blockIdx.x * 32;

  // ---- stage x tile (32 x 256 f32) into LDS as bf16, non-temporal ----
  for (int i = 0; i < 32; ++i) {
    int idx  = t + (i << 8);
    int r    = idx >> 8;
    int kcol = idx & 255;
    float v = __builtin_nontemporal_load(&x[(row_base + r) * CC + kcol]);
    aLDS[r * LDS_STRIDE + kcol] = (unsigned short)f2bf_bits(v);
  }
  __syncthreads();

  // constant bias per output column: b_in[n] + b[n]
  float binv[4];
  #pragma unroll
  for (int j = 0; j < 4; ++j) {
    int n = (wave * 4 + j) * 16 + m;
    binv[j] = b_in[n] + brec[n];
  }

  const v8f vzero = {0.f, 0.f, 0.f, 0.f, 0.f, 0.f, 0.f, 0.f};

  // ---- in_proj: cb = x @ W_in^T  (kt = 0..7) ----
  v8f cb[2][4];
  #pragma unroll
  for (int mt = 0; mt < 2; ++mt)
    #pragma unroll
    for (int j = 0; j < 4; ++j) cb[mt][j] = vzero;

  #pragma unroll 1
  for (int kt = 0; kt < 8; ++kt) {
    v16bf A0 = load_afrag_lds(aLDS, (m)      * (LDS_STRIDE * 2) + kt * 64 + khalf * 32);
    v16bf A1 = load_afrag_lds(aLDS, (16 + m) * (LDS_STRIDE * 2) + kt * 64 + khalf * 32);
    #pragma unroll
    for (int j = 0; j < 4; ++j) {
      int nt = wave * 4 + j;
      v16bf B = load_bfrag(Winf + (kt * 32 + nt) * 256, lane);
      cb[0][j] = WMMA_BF16(A0, B, cb[0][j]);
      cb[1][j] = WMMA_BF16(A1, B, cb[1][j]);
    }
  }
  #pragma unroll
  for (int mt = 0; mt < 2; ++mt)
    #pragma unroll
    for (int j = 0; j < 4; ++j)
      #pragma unroll
      for (int g = 0; g < 8; ++g) cb[mt][j][g] += binv[j];

  // ---- iteration 1: a = tanh(cb) (a0 = 0) ----
  __syncthreads();   // all waves done reading x fragments
  #pragma unroll
  for (int mt = 0; mt < 2; ++mt)
    #pragma unroll
    for (int j = 0; j < 4; ++j) {
      int nt = wave * 4 + j;
      #pragma unroll
      for (int g = 0; g < 8; ++g) {
        float v = fast_tanh(cb[mt][j][g]);
        aLDS[(mt * 16 + g + 8 * khalf) * LDS_STRIDE + nt * 16 + m] =
            (unsigned short)f2bf_bits(v);
      }
    }
  __syncthreads();

  // ---- iterations 2..15: a = tanh(a @ Ws + cb) ----
  #pragma unroll 1
  for (int it = 0; it < KITER - 1; ++it) {
    v8f acc[2][4];
    #pragma unroll
    for (int mt = 0; mt < 2; ++mt)
      #pragma unroll
      for (int j = 0; j < 4; ++j) acc[mt][j] = cb[mt][j];

    #pragma unroll 1
    for (int kt = 0; kt < 16; ++kt) {
      v16bf A0 = load_afrag_lds(aLDS, (m)      * (LDS_STRIDE * 2) + kt * 64 + khalf * 32);
      v16bf A1 = load_afrag_lds(aLDS, (16 + m) * (LDS_STRIDE * 2) + kt * 64 + khalf * 32);
      #pragma unroll
      for (int j = 0; j < 4; ++j) {
        int nt = wave * 4 + j;
        v16bf B = load_bfrag(Wsf + (kt * 32 + nt) * 256, lane);
        acc[0][j] = WMMA_BF16(A0, B, acc[0][j]);
        acc[1][j] = WMMA_BF16(A1, B, acc[1][j]);
      }
    }
    __syncthreads();   // all reads of a done before overwrite
    #pragma unroll
    for (int mt = 0; mt < 2; ++mt)
      #pragma unroll
      for (int j = 0; j < 4; ++j) {
        int nt = wave * 4 + j;
        #pragma unroll
        for (int g = 0; g < 8; ++g) {
          float v = fast_tanh(acc[mt][j][g]);
          aLDS[(mt * 16 + g + 8 * khalf) * LDS_STRIDE + nt * 16 + m] =
              (unsigned short)f2bf_bits(v);
        }
      }
    __syncthreads();
  }

  // ---- out_proj: y = a @ W_out^T + b_out  (n-tiles 0..15, kt 0..15) ----
  v8f oacc[2][2];
  #pragma unroll
  for (int mt = 0; mt < 2; ++mt)
    #pragma unroll
    for (int j = 0; j < 2; ++j) oacc[mt][j] = vzero;

  #pragma unroll 1
  for (int kt = 0; kt < 16; ++kt) {
    v16bf A0 = load_afrag_lds(aLDS, (m)      * (LDS_STRIDE * 2) + kt * 64 + khalf * 32);
    v16bf A1 = load_afrag_lds(aLDS, (16 + m) * (LDS_STRIDE * 2) + kt * 64 + khalf * 32);
    #pragma unroll
    for (int j = 0; j < 2; ++j) {
      int nt = wave * 2 + j;
      v16bf B = load_bfrag(Woutf + (kt * 16 + nt) * 256, lane);
      oacc[0][j] = WMMA_BF16(A0, B, oacc[0][j]);
      oacc[1][j] = WMMA_BF16(A1, B, oacc[1][j]);
    }
  }

  #pragma unroll
  for (int j = 0; j < 2; ++j) {
    int n = (wave * 2 + j) * 16 + m;
    float bo = b_out[n];
    #pragma unroll
    for (int mt = 0; mt < 2; ++mt)
      #pragma unroll
      for (int g = 0; g < 8; ++g) {
        float v = oacc[mt][j][g] + bo;
        __builtin_nontemporal_store(
            v, &y[(row_base + mt * 16 + g + 8 * khalf) * CC + n]);
      }
  }
}

extern "C" void kernel_launch(void* const* d_in, const int* in_sizes, int n_in,
                              void* d_out, int out_size, void* d_ws, size_t ws_size,
                              hipStream_t stream) {
  const float* x     = (const float*)d_in[0];
  const float* W_in  = (const float*)d_in[1];
  const float* b_in  = (const float*)d_in[2];
  const float* W     = (const float*)d_in[3];
  const float* b     = (const float*)d_in[4];
  const float* W_out = (const float*)d_in[5];
  const float* b_out = (const float*)d_in[6];
  float* y = (float*)d_out;

  // workspace: pre-swizzled bf16 weight fragments (1 MB total)
  unsigned int* ws    = (unsigned int*)d_ws;
  unsigned int* Wsf   = ws;            // 16*32 frags * 256 dwords = 131072 (512 KB)
  unsigned int* Winf  = ws + 131072;   //  8*32 frags * 256 dwords =  65536 (256 KB)
  unsigned int* Woutf = ws + 196608;   // 16*16 frags * 256 dwords =  65536 (256 KB)

  prep_ws  <<<512, 256, 0, stream>>>(W,     Wsf);
  prep_win <<<256, 256, 0, stream>>>(W_in,  Winf);
  prep_wout<<<256, 256, 0, stream>>>(W_out, Woutf);

  int rows = in_sizes[0] / CC;   // 65536
  int wgs  = rows / 32;          // 2048 workgroups, 32 rows each
  attractor_main<<<wgs, 256, 0, stream>>>(x, b_in, b, b_out, Wsf, Winf, Woutf, y);
}
